// Head_55216099558126
// MI455X (gfx1250) — matrix operand order repaired and checked by reading, not compiled, not run on hardware
//
#include <hip/hip_runtime.h>
#include <hip/hip_bf16.h>

// ---------------------------------------------------------------------------
// Causal self-attention head for MI455X (gfx1250).
//   k = x@Wk^T + bk ; q = x@Wq^T + bq ; v = x@Wv^T + bv
//   S = causal(q@k^T * C^-0.5) ; P = softmax(S) ; y = P@v
// Single WMMA GEMM kernel (C[m,n] = sum_k A[m,k]*B[n,k], A/B row-major bf16):
//   A tile staged by the Tensor Data Mover (tensor_load_to_lds, TENSORcnt,
//   hardware LDS row padding), B tile by GLOBAL_LOAD_ASYNC_TO_LDS_B128
//   (ASYNCcnt), both double-buffered; compute via v_wmma_f32_16x16x32_bf16
//   fed from ds_load_b128 fragments.
// ---------------------------------------------------------------------------

typedef __bf16 bf16;
typedef __attribute__((ext_vector_type(16))) __bf16 bf16x16;
typedef __attribute__((ext_vector_type(8)))  __bf16 bf16x8;
typedef __attribute__((ext_vector_type(8)))  float  f32x8;
typedef unsigned int u32x4 __attribute__((ext_vector_type(4)));
typedef int          i32x4 __attribute__((ext_vector_type(4)));
typedef int          i32x8 __attribute__((ext_vector_type(8)));

constexpr int kC = 1024;           // embedding dim
constexpr int kT = 2048;           // sequence length
constexpr int kB = 4;              // batch
constexpr int kM_ALL = kB * kT;    // 8192 tokens

constexpr int KC  = 64;            // K chunk staged per stage
constexpr int KAP = KC + 8;        // padded LDS row stride (144B) to spread banks

#if __has_builtin(__builtin_amdgcn_tensor_load_to_lds)
#define USE_TDM 1
#else
#define USE_TDM 0
#endif

// -------------------------------------------------------------------- helpers

// Combine two 16B chunks into one 16-element bf16 fragment (8 VGPRs).
__device__ __forceinline__ bf16x16 frag2(const bf16* plo, const bf16* phi) {
  bf16x8 lo = *(const bf16x8*)plo;
  bf16x8 hi = *(const bf16x8*)phi;
  return __builtin_shufflevector(lo, hi, 0, 1, 2, 3, 4, 5, 6, 7,
                                 8, 9, 10, 11, 12, 13, 14, 15);
}

// Generic LDS pointer -> 32-bit LDS byte offset (low 32 bits of the shared
// aperture address are the LDS offset).
__device__ __forceinline__ unsigned int lds_off(const void* p) {
  return (unsigned int)(unsigned long long)p;
}

// One 16-byte async DMA: global -> LDS, tracked by ASYNCcnt, no VGPR data.
__device__ __forceinline__ void async_ld16(unsigned int dst_lds,
                                           const bf16* src_global) {
  asm volatile("global_load_async_to_lds_b128 %0, %1, off"
               :: "v"(dst_lds),
                  "v"((unsigned long long)src_global)
               : "memory");
}

#if USE_TDM
// Tensor Data Mover: 2D bf16 tile (tile_w x tile_h, row stride stride_elems)
// from global to LDS at lds_addr, with hardware LDS padding of 16B after each
// 128B row (pad_interval code 4 = 32 DWORDs, pad_amount code 3 = 4 DWORDs),
// matching the KAP=72-element padded LDS rows. tensor_dim == tile_dim, so no
// OOB clipping; tensor_dim0_stride carries the real row pitch.
__device__ __forceinline__ void tdm_load_2d(unsigned int lds_addr,
                                            const bf16* gsrc,
                                            unsigned int tile_w,
                                            unsigned int tile_h,
                                            unsigned int stride_elems) {
  unsigned long long ga = (unsigned long long)gsrc;
  u32x4 g0;
  g0[0] = 1u;                                    // count=1 (valid), user mode
  g0[1] = lds_addr;                              // D#.lds_addr (bytes)
  g0[2] = (unsigned int)(ga & 0xffffffffu);      // global_addr[31:0]
  g0[3] = (unsigned int)((ga >> 32) & 0x01ffffffu)
        | (2u << 30);                            // global_addr[56:32] | type=2
  i32x8 g1;
  g1[0] = (1 << 16)                              // data_size = 2 bytes
        | (1 << 20)                              // pad_enable
        | (4 << 22)                              // pad_interval: 32 DWORDs
        | (3 << 25);                             // pad_amount: 4 DWORDs
  g1[1] = (int)(tile_w << 16);                   // tensor_dim0[15:0]
  g1[2] = (int)(tile_h << 16);                   // dim0[31:16]=0 | dim1[15:0]
  g1[3] = (int)(tile_w << 16);                   // dim1[31:16]=0 | tile_dim0
  g1[4] = (int)tile_h;                           // tile_dim1 | tile_dim2=0
  g1[5] = (int)stride_elems;                     // tensor_dim0_stride[31:0]
  g1[6] = 0;                                     // stride[47:32] | dim1_stride lo
  g1[7] = 0;
  i32x4 z4 = {0, 0, 0, 0};
  i32x8 z8 = {0, 0, 0, 0, 0, 0, 0, 0};
  // amdgpu-toolchain (clang-23) 6-arg form: (g0, g1, g2, g3, g4, cpol)
  __builtin_amdgcn_tensor_load_to_lds(g0, g1, z4, z4, z8, 0);
}
#endif

// ------------------------------------------------------------------ conversion

__global__ __launch_bounds__(256) void cvt_f32_to_bf16(
    const float* __restrict__ src, bf16* __restrict__ dst, int n) {
  int i = blockIdx.x * 256 + threadIdx.x;
  if (i < n) dst[i] = (bf16)src[i];
}

// ------------------------------------------------------------------- WMMA GEMM
// MODE 0: out bf16 = acc + bias[n]                  (q/k projection)
// MODE 1: out bf16 transposed per-batch vT[b][c][t] (v projection)
// MODE 2: out f32  = causal-masked acc * scale      (attention scores)
// MODE 3: out f32  = acc                            (P @ V -> y)
// Block: 256 threads = 8 waves (4x2), block tile 128(M) x 64(N),
// wave tile 32x32 = 2x2 of 16x16 WMMA accumulators, K staged in LDS chunks.

template <int MODE>
__global__ __launch_bounds__(256) void wmma_gemm_nt(
    const bf16* __restrict__ A, const bf16* __restrict__ B,
    const float* __restrict__ bias, void* __restrict__ out,
    int K, int lda, int ldb, int ldo, float scale) {
  const int tid  = threadIdx.x;
  const int lane = tid & 31;
  const int wv   = tid >> 5;    // wave id 0..7
  const int wm   = wv >> 1;     // 0..3 (M direction)
  const int wn   = wv & 1;      // 0..1 (N direction)
  const int hl   = lane >> 4;   // lane half (K split)
  const int lr   = lane & 15;

  const int blockM = blockIdx.y * 128;
  const int blockN = blockIdx.x * 64;

  if constexpr (MODE == 2) {
    // Tile entirely above the diagonal: fill -inf, skip the GEMM.
    if (blockN > blockM + 127) {
      float* o = (float*)out;
      for (int i = tid; i < 128 * 64; i += 256) {
        int r = i >> 6, c = i & 63;
        o[(size_t)(blockM + r) * ldo + (blockN + c)] = -__builtin_inff();
      }
      return;
    }
  }

  __shared__ __align__(16) bf16 As[2][128][KAP];
  __shared__ __align__(16) bf16 Bs[2][64][KAP];

  const bf16* Ag = A + (size_t)blockM * lda;   // block's A rows
  const bf16* Bg = B + (size_t)blockN * ldb;   // block's B rows

  const int rsub = lane >> 3;          // 0..3  (row within 4-row group)
  const int csub = (lane & 7) * 8;     // element column of this lane's 16B

  // Stage one K-chunk: A tile (128 x KC) via one TDM descriptor issued by
  // wave 0; B tile (64 x KC) via per-lane async DMA, 2 issues per wave
  // (each issue = 32 lanes x 16B = 4 rows).
  auto stage = [&](int buf, int kc) {
#if USE_TDM
    if (wv == 0) {
      tdm_load_2d(lds_off(&As[buf][0][0]), Ag + kc, KC, 128, (unsigned)lda);
    }
#else
    #pragma unroll
    for (int j = 0; j < 4; ++j) {
      int r = (wv * 4 + j) * 4 + rsub;           // 0..127
      async_ld16(lds_off(&As[buf][r][csub]),
                 Ag + (size_t)r * lda + kc + csub);
    }
#endif
    #pragma unroll
    for (int j = 0; j < 2; ++j) {
      int r = (wv * 2 + j) * 4 + rsub;           // 0..63
      async_ld16(lds_off(&Bs[buf][r][csub]),
                 Bg + (size_t)r * ldb + kc + csub);
    }
  };

  f32x8 acc00 = {}, acc01 = {}, acc10 = {}, acc11 = {};

  const int nch = K / KC;
  stage(0, 0);

  for (int c = 0; c < nch; ++c) {
    const int buf = c & 1;
    if (c + 1 < nch) {
      stage(buf ^ 1, (c + 1) * KC);
      // Per-wave in-order retirement: with stage c+1 just issued, waiting
      // until only the new issues remain guarantees stage c completed.
#if USE_TDM
      if (wv == 0) __builtin_amdgcn_s_wait_tensorcnt(1);
      asm volatile("s_wait_asynccnt 0x2" ::: "memory");
#else
      asm volatile("s_wait_asynccnt 0x6" ::: "memory");
#endif
    } else {
#if USE_TDM
      if (wv == 0) __builtin_amdgcn_s_wait_tensorcnt(0);
#endif
      asm volatile("s_wait_asynccnt 0x0" ::: "memory");
    }
    __syncthreads();

    #pragma unroll
    for (int ks = 0; ks < KC; ks += 32) {
      const bf16* pa0 = &As[buf][wm * 32 + lr][ks];
      const bf16* pa1 = &As[buf][wm * 32 + 16 + lr][ks];
      const bf16* pb0 = &Bs[buf][wn * 32 + lr][ks + 8 * hl];
      const bf16* pb1 = &Bs[buf][wn * 32 + 16 + lr][ks + 8 * hl];
      // A frag: lane half 0 -> K {0..7,16..23}; half 1 -> K {8..15,24..31}
      bf16x16 a0 = frag2(pa0 + 8 * hl, pa0 + 16 + 8 * hl);
      bf16x16 a1 = frag2(pa1 + 8 * hl, pa1 + 16 + 8 * hl);
      // B frag: lane half 0 -> K {0..7,16..23}; half 1 -> K {8..15,24..31}
      bf16x16 b0 = frag2(pb0, pb0 + 16);
      bf16x16 b1 = frag2(pb1, pb1 + 16);
      acc00 = __builtin_amdgcn_wmma_f32_16x16x32_bf16(false, a0, false, b0,
                                                      (short)0, acc00, false, false);
      acc01 = __builtin_amdgcn_wmma_f32_16x16x32_bf16(false, a0, false, b1,
                                                      (short)0, acc01, false, false);
      acc10 = __builtin_amdgcn_wmma_f32_16x16x32_bf16(false, a1, false, b0,
                                                      (short)0, acc10, false, false);
      acc11 = __builtin_amdgcn_wmma_f32_16x16x32_bf16(false, a1, false, b1,
                                                      (short)0, acc11, false, false);
    }
    __syncthreads();   // stage c done for every wave before buf is re-staged
  }

  const int m0 = blockM + wm * 32;
  const int n0 = blockN + wn * 32;

  // Epilogue. D layout (16x16 f32): VGPR e -> row m0 + 8*hl + e, col n0 + lr.
  f32x8 accs[2][2] = {{acc00, acc01}, {acc10, acc11}};
  #pragma unroll
  for (int sm = 0; sm < 2; ++sm) {
    #pragma unroll
    for (int sn = 0; sn < 2; ++sn) {
      const int col = n0 + sn * 16 + lr;
      float badd = 0.0f;
      if constexpr (MODE <= 1) badd = bias[col];
      #pragma unroll
      for (int e = 0; e < 8; ++e) {
        const int row = m0 + sm * 16 + hl * 8 + e;
        float v = accs[sm][sn][e] + badd;
        if constexpr (MODE == 0) {
          ((bf16*)out)[(size_t)row * ldo + col] = (bf16)v;
        } else if constexpr (MODE == 1) {
          const int bi = row / kT, t = row % kT;  // vT[b][c][t]
          ((bf16*)out)[(size_t)bi * kC * kT + (size_t)col * kT + t] = (bf16)v;
        } else if constexpr (MODE == 2) {
          float s = (col <= row) ? v * scale : -__builtin_inff();
          ((float*)out)[(size_t)row * ldo + col] = s;
        } else {
          ((float*)out)[(size_t)row * ldo + col] = v;
        }
      }
    }
  }
}

// --------------------------------------------------------------------- softmax
// One block (8 waves) per score row; wave32 shuffle + LDS cross-wave reduce.

__global__ __launch_bounds__(256) void softmax_rows(
    const float* __restrict__ S, bf16* __restrict__ P) {
  const int row = blockIdx.x;
  const float* s = S + (size_t)row * kT;
  bf16* p = P + (size_t)row * kT;
  __shared__ float redMax[8];
  __shared__ float redSum[8];
  const int tid = threadIdx.x;

  float mx = -__builtin_inff();
  for (int i = tid; i < kT; i += 256) mx = fmaxf(mx, s[i]);
  #pragma unroll
  for (int o = 16; o > 0; o >>= 1) mx = fmaxf(mx, __shfl_xor(mx, o, 32));
  if ((tid & 31) == 0) redMax[tid >> 5] = mx;
  __syncthreads();
  float m = redMax[0];
  #pragma unroll
  for (int i = 1; i < 8; ++i) m = fmaxf(m, redMax[i]);

  float sum = 0.0f;
  for (int i = tid; i < kT; i += 256) sum += __expf(s[i] - m);
  #pragma unroll
  for (int o = 16; o > 0; o >>= 1) sum += __shfl_xor(sum, o, 32);
  if ((tid & 31) == 0) redSum[tid >> 5] = sum;
  __syncthreads();
  float tot = 0.0f;
  #pragma unroll
  for (int i = 0; i < 8; ++i) tot += redSum[i];

  const float inv = 1.0f / tot;
  for (int i = tid; i < kT; i += 256) p[i] = (bf16)(__expf(s[i] - m) * inv);
}

// ---------------------------------------------------------------------- launch

extern "C" void kernel_launch(void* const* d_in, const int* in_sizes, int n_in,
                              void* d_out, int out_size, void* d_ws, size_t ws_size,
                              hipStream_t stream) {
  const float* x  = (const float*)d_in[0];
  const float* Wk = (const float*)d_in[1];
  const float* bk = (const float*)d_in[2];
  const float* Wq = (const float*)d_in[3];
  const float* bq = (const float*)d_in[4];
  const float* Wv = (const float*)d_in[5];
  const float* bv = (const float*)d_in[6];
  float* y = (float*)d_out;

  char* ws = (char*)d_ws;
  size_t off = 0;
  auto wsalloc = [&](size_t bytes) -> void* {
    void* p = ws + off;
    off += (bytes + 255) & ~(size_t)255;
    return p;
  };
  bf16*  xb  = (bf16*)wsalloc((size_t)kM_ALL * kC * 2);
  bf16*  wkb = (bf16*)wsalloc((size_t)kC * kC * 2);
  bf16*  wqb = (bf16*)wsalloc((size_t)kC * kC * 2);
  bf16*  wvb = (bf16*)wsalloc((size_t)kC * kC * 2);
  bf16*  qb  = (bf16*)wsalloc((size_t)kM_ALL * kC * 2);
  bf16*  kbf = (bf16*)wsalloc((size_t)kM_ALL * kC * 2);
  bf16*  vTb = (bf16*)wsalloc((size_t)kB * kC * kT * 2);
  float* S   = (float*)wsalloc((size_t)kT * kT * 4);   // one batch at a time
  bf16*  P   = (bf16*)wsalloc((size_t)kT * kT * 2);

  // 1) f32 -> bf16 conversions
  {
    int n = kM_ALL * kC;
    cvt_f32_to_bf16<<<(n + 255) / 256, 256, 0, stream>>>(x, xb, n);
    n = kC * kC;
    cvt_f32_to_bf16<<<(n + 255) / 256, 256, 0, stream>>>(Wk, wkb, n);
    cvt_f32_to_bf16<<<(n + 255) / 256, 256, 0, stream>>>(Wq, wqb, n);
    cvt_f32_to_bf16<<<(n + 255) / 256, 256, 0, stream>>>(Wv, wvb, n);
  }

  dim3 blk(256);

  // 2) QKV projections: [8192,1024] x [1024,1024]^T
  dim3 g_qkv(kC / 64, kM_ALL / 128);  // (16, 64)
  wmma_gemm_nt<0><<<g_qkv, blk, 0, stream>>>(xb, wqb, bq, qb, kC, kC, kC, kC, 1.0f);
  wmma_gemm_nt<0><<<g_qkv, blk, 0, stream>>>(xb, wkb, bk, kbf, kC, kC, kC, kC, 1.0f);
  wmma_gemm_nt<1><<<g_qkv, blk, 0, stream>>>(xb, wvb, bv, vTb, kC, kC, kC, 0, 1.0f);

  // 3) Per-batch attention (stream-ordered reuse of S/P scratch)
  const float scale = 1.0f / 32.0f;   // C^-0.5, C = 1024
  dim3 g_sc(kT / 64, kT / 128);       // (32, 16)
  dim3 g_pv(kC / 64, kT / 128);       // (16, 16)
  for (int b = 0; b < kB; ++b) {
    const bf16* qB  = qb  + (size_t)b * kT * kC;
    const bf16* kB_ = kbf + (size_t)b * kT * kC;
    const bf16* vTB = vTb + (size_t)b * kC * kT;
    float* yB = y + (size_t)b * kT * kC;
    wmma_gemm_nt<2><<<g_sc, blk, 0, stream>>>(qB, kB_, nullptr, S,
                                              kC, kC, kC, kT, scale);
    softmax_rows<<<kT, 256, 0, stream>>>(S, P);
    wmma_gemm_nt<3><<<g_pv, blk, 0, stream>>>(P, vTB, nullptr, yB,
                                              kT, kT, kT, kC, 1.0f);
  }
}
